// PERSON_HEAD_22462678958361
// MI455X (gfx1250) — matrix operand order repaired and checked by reading, not compiled, not run on hardware
//
#include <hip/hip_runtime.h>
#include <math.h>

typedef __bf16 bf16_t;
typedef bf16_t v16bf __attribute__((ext_vector_type(16)));
typedef float  v8f   __attribute__((ext_vector_type(8)));

#define HH 56
#define WW 56
#define NA 9
#define KCAND 1024
#define MAXDET 100
#define CONF_THR 0.3f
#define NMS_IOU  0.2f

__device__ __forceinline__ unsigned short f2bf(float f) {
  unsigned int u = __float_as_uint(f);
  u += 0x7FFFu + ((u >> 16) & 1u);   // round-to-nearest-even
  return (unsigned short)(u >> 16);
}
__device__ __forceinline__ float bf2f(unsigned short h) {
  return __uint_as_float(((unsigned int)h) << 16);
}
__device__ __forceinline__ float sigmf(float x) { return 1.f / (1.f + expf(-x)); }

// ---------------------------------------------------------------------------
// Prologue: pack conv weights into WMMA-B-fragment order, split hi/lo bf16.
// Layout: [T = COUT/16][S = 9*CIN/32][lane 0..31][v 0..7] packed dwords.
// k ordering: k = tap*CIN + c (tap-major) so each 32-step is one 3x3 tap.
// B-frag as packed here (16-bit, 32x16, contiguous k, matched by A): lane L -> N = L&15, half = L>>4, VGPR v holds
// K = half*16 + 2v (low 16 bits) and K+1 (high 16 bits).
// Also precomputes per-channel BN fold: sc = g*rsqrt(v+eps),
// sh = (conv_bias - mean)*sc + beta.
// ---------------------------------------------------------------------------
__global__ void pack_weights(const float* __restrict__ w, const float* __restrict__ cb,
                             const float* __restrict__ g, const float* __restrict__ be,
                             const float* __restrict__ mean, const float* __restrict__ var,
                             unsigned int* __restrict__ whi, unsigned int* __restrict__ wlo,
                             float* __restrict__ scsh, int CIN, int COUT) {
  int S = 9 * CIN / 32;
  int total = (COUT / 16) * S * 256;
  int gsz = gridDim.x * blockDim.x;
  int gtid = blockIdx.x * blockDim.x + threadIdx.x;
  for (int i = gtid; i < total; i += gsz) {
    int vv = i & 7;
    int lane = (i >> 3) & 31;
    int s = (i >> 8) % S;
    int t = i / (S * 256);
    int n = lane & 15, half = lane >> 4;
    int k = s * 32 + half * 16 + 2 * vv;
    int tap = k / CIN;
    int c = k - tap * CIN;
    int cout = t * 16 + n;
    float f0 = w[((size_t)cout * CIN + c) * 9 + tap];
    float f1 = w[((size_t)cout * CIN + c + 1) * 9 + tap];
    unsigned short h0 = f2bf(f0), h1 = f2bf(f1);
    whi[i] = ((unsigned int)h1 << 16) | h0;
    unsigned short l0 = f2bf(f0 - bf2f(h0)), l1 = f2bf(f1 - bf2f(h1));
    wlo[i] = ((unsigned int)l1 << 16) | l0;
  }
  for (int c = gtid; c < COUT; c += gsz) {
    float sc = g[c] * rsqrtf(var[c] + 1e-5f);
    scsh[c] = sc;
    scsh[COUT + c] = (cb[c] - mean[c]) * sc + be[c];
  }
}

// ---------------------------------------------------------------------------
// 3x3 conv + BN + ReLU via implicit GEMM on v_wmma_f32_16x16x32_bf16,
// split-bf16 (hi+lo) x3 products for ~fp32 precision, f32 accumulate.
// One workgroup = one (b, y, 16-pixel x-tile); wave w owns cout tile w.
// Input patch (3 rows x 18 cols x CIN) staged in LDS pre-split into bf16
// hi/lo channel-pairs so the A fragment is pure ds_load_b32 gathers.
// Epilogue transposes through LDS for contiguous global stores.
// ---------------------------------------------------------------------------
template <int CIN, int COUT>
__global__ __launch_bounds__(32 * (COUT / 16)) void conv3x3_bn_relu(
    const float* __restrict__ in, const unsigned int* __restrict__ wb_hi,
    const unsigned int* __restrict__ wb_lo, const float* __restrict__ scsh,
    float* __restrict__ out) {
  constexpr int NW = COUT / 16;     // waves per block
  constexpr int S = 9 * CIN / 32;   // K steps
  constexpr int CP = CIN / 2;       // channel pairs

  __shared__ unsigned int lhi[CP * 3 * 18];
  __shared__ unsigned int llo[CP * 3 * 18];
  __shared__ float ltr[NW][16][17];

  int bid = blockIdx.x;
  int tx = bid & 3;
  int y = (bid >> 2) % HH;
  int b = bid / (4 * HH);
  int x0 = tx * 16;
  int tid = threadIdx.x;
  const float* inb = in + (size_t)b * CIN * HH * WW;

  // Stage halo patch, pre-split to bf16 hi/lo pairs (pair = consecutive cin).
  for (int i = tid; i < CP * 3 * 18; i += 32 * NW) {
    int xx = i % 18;
    int t2 = i / 18;
    int row = t2 % 3;
    int cp = t2 / 3;
    int gy = y + row - 1;
    int gx = x0 + xx - 1;
    float f0 = 0.f, f1 = 0.f;
    if ((unsigned)gy < (unsigned)HH && (unsigned)gx < (unsigned)WW) {
      const float* p = inb + ((size_t)(2 * cp) * HH + gy) * WW + gx;
      f0 = p[0];
      f1 = p[(size_t)HH * WW];
    }
    unsigned short h0 = f2bf(f0), h1 = f2bf(f1);
    lhi[i] = ((unsigned int)h1 << 16) | h0;
    unsigned short l0 = f2bf(f0 - bf2f(h0)), l1 = f2bf(f1 - bf2f(h1));
    llo[i] = ((unsigned int)l1 << 16) | l0;
  }
  __syncthreads();

  int wid = tid >> 5, lane = tid & 31;
  int m = lane & 15, half = lane >> 4;
  v8f acc = {};
  const unsigned int* bh = wb_hi + (size_t)wid * S * 256 + lane * 8;
  const unsigned int* bl = wb_lo + (size_t)wid * S * 256 + lane * 8;

  for (int s = 0; s < S; ++s) {
    int tap = s / (CIN / 32);
    int c0p = (s % (CIN / 32)) * 16;
    int dy = tap / 3, dx = tap - 3 * dy;

    // A fragment (16-bit 16x32): lane m = pixel; v0..7 -> K pairs half*8+v.
    union { unsigned int u[8]; v16bf v; } ahi, alo;
#pragma unroll
    for (int vv = 0; vv < 8; ++vv) {
      int cp = c0p + half * 8 + vv;
      int idx = (cp * 3 + dy) * 18 + (dx + m);
      ahi.u[vv] = lhi[idx];
      alo.u[vv] = llo[idx];
    }
    // B fragment: two b128 loads each from packed, L2-resident weights.
    union { uint4 q[2]; v16bf v; } bhi, blo;
    const uint4* ph = reinterpret_cast<const uint4*>(bh + (size_t)s * 256);
    const uint4* pl = reinterpret_cast<const uint4*>(bl + (size_t)s * 256);
    bhi.q[0] = ph[0]; bhi.q[1] = ph[1];
    blo.q[0] = pl[0]; blo.q[1] = pl[1];

    acc = __builtin_amdgcn_wmma_f32_16x16x32_bf16(false, ahi.v, false, bhi.v,
                                                  (short)0, acc, false, false);
    acc = __builtin_amdgcn_wmma_f32_16x16x32_bf16(false, ahi.v, false, blo.v,
                                                  (short)0, acc, false, false);
    acc = __builtin_amdgcn_wmma_f32_16x16x32_bf16(false, alo.v, false, bhi.v,
                                                  (short)0, acc, false, false);
  }

  // Epilogue: BN + ReLU, transpose via LDS for coalesced stores.
  float sc = scsh[wid * 16 + m];
  float sh = scsh[COUT + wid * 16 + m];
#pragma unroll
  for (int r = 0; r < 8; ++r) {
    int px = r + 8 * half;   // D: VGPR r -> M = r + 8*half
    float v = acc[r] * sc + sh;
    ltr[wid][m][px] = v > 0.f ? v : 0.f;
  }
  __syncthreads();
  int row = lane >> 1, hp = lane & 1;
  int coutg = wid * 16 + row;
  float* op = out + (((size_t)b * COUT + coutg) * HH + y) * WW;
#pragma unroll
  for (int j = 0; j < 8; ++j) {
    int px = hp * 8 + j;
    int x = x0 + px;
    if (x < WW) op[x] = ltr[wid][row][px];
  }
}

// ---------------------------------------------------------------------------
// 1x1 head conv (64 -> 54) + YOLO decode. One thread per (b, pixel, anchor).
// Weights staged in LDS (13.8KB).
// ---------------------------------------------------------------------------
__global__ void head_decode(const float* __restrict__ x2, const float* __restrict__ wo,
                            const float* __restrict__ bo, float* __restrict__ boxes,
                            float* __restrict__ scores, int total) {
  __shared__ float sw[54 * 64];
  __shared__ float sb[54];
  for (int i = threadIdx.x; i < 54 * 64; i += blockDim.x) sw[i] = wo[i];
  for (int i = threadIdx.x; i < 54; i += blockDim.x) sb[i] = bo[i];
  __syncthreads();
  int gsz = gridDim.x * blockDim.x;
  for (int i = blockIdx.x * blockDim.x + threadIdx.x; i < total; i += gsz) {
    int a = i % NA;
    int pix = (i / NA) % (HH * WW);
    int b = i / (NA * HH * WW);
    int x = pix % WW, yy = pix / WW;
    const float* xp = x2 + (size_t)b * 64 * HH * WW + pix;
    float xv[64];
#pragma unroll
    for (int c = 0; c < 64; ++c) xv[c] = xp[(size_t)c * HH * WW];
    float t[6];
    for (int f = 0; f < 6; ++f) {
      int co = a * 6 + f;
      float acc = sb[co];
      const float* wr = &sw[co * 64];
#pragma unroll
      for (int c = 0; c < 64; ++c) acc += wr[c] * xv[c];
      t[f] = acc;
    }
    const float sz[3] = {32.f, 64.f, 128.f};
    const float ar[3] = {0.5f, 1.f, 2.f};
    float s_ = sz[a / 3], rr = ar[a % 3];
    float aw = s_ * rr / 224.f;
    float ah = s_ / rr / 224.f;
    float px = (sigmf(t[0]) + (float)x) / (float)WW;
    float py = (sigmf(t[1]) + (float)yy) / (float)HH;
    float pw = expf(t[2]) * aw;
    float ph = expf(t[3]) * ah;
    size_t o = (size_t)b * (HH * WW * NA) + (size_t)pix * NA + a;
    boxes[o * 4 + 0] = px;
    boxes[o * 4 + 1] = py;
    boxes[o * 4 + 2] = pw;
    boxes[o * 4 + 3] = ph;
    scores[o] = sigmf(t[4]);
  }
}

// ---------------------------------------------------------------------------
// Top-1024 per image: 32-round bitwise radix descent on float bit patterns
// (scores are positive so uint order == float order), then LDS-atomic compact.
// ---------------------------------------------------------------------------
__global__ void topk1024(const float* __restrict__ scores, int* __restrict__ cidx,
                         float* __restrict__ cscore) {
  constexpr int N = HH * WW * NA;
  __shared__ int red[256];
  __shared__ int s_cnt1, s_cnt2;
  int b = blockIdx.x;
  const float* sc = scores + (size_t)b * N;
  int tid = threadIdx.x;
  unsigned int prefix = 0;
  for (int bit = 31; bit >= 0; --bit) {
    unsigned int test = prefix | (1u << bit);
    int c = 0;
    for (int i = tid; i < N; i += blockDim.x)
      c += (__float_as_uint(sc[i]) >= test);
    red[tid] = c;
    __syncthreads();
    for (int off = 128; off > 0; off >>= 1) {
      if (tid < off) red[tid] += red[tid + off];
      __syncthreads();
    }
    int tot = red[0];
    __syncthreads();
    if (tot >= KCAND) prefix = test;
  }
  // prefix == 1024th-largest key
  if (tid == 0) s_cnt1 = 0;
  __syncthreads();
  for (int i = tid; i < N; i += blockDim.x) {
    if (__float_as_uint(sc[i]) > prefix) {
      int pos = atomicAdd(&s_cnt1, 1);
      cidx[b * KCAND + pos] = i;
      cscore[b * KCAND + pos] = sc[i];
    }
  }
  __syncthreads();
  if (tid == 0) s_cnt2 = s_cnt1;
  __syncthreads();
  for (int i = tid; i < N; i += blockDim.x) {
    if (__float_as_uint(sc[i]) == prefix) {
      int pos = atomicAdd(&s_cnt2, 1);
      if (pos < KCAND) {
        cidx[b * KCAND + pos] = i;
        cscore[b * KCAND + pos] = sc[i];
      }
    }
  }
}

// ---------------------------------------------------------------------------
// Greedy NMS: one 1024-thread workgroup per image. Argmax tree (ties ->
// lowest index, matching jnp.argmax), exact reference IoU math.
// ---------------------------------------------------------------------------
__global__ __launch_bounds__(1024) void nms_kernel(
    const float* __restrict__ boxes, const int* __restrict__ cidx,
    const float* __restrict__ cscore, float* __restrict__ out) {
  constexpr int N = HH * WW * NA;
  __shared__ float s_s[KCAND], s_x1[KCAND], s_y1[KCAND], s_x2[KCAND], s_y2[KCAND];
  __shared__ float s_ar[KCAND], s_cx[KCAND], s_cy[KCAND], s_w[KCAND], s_h[KCAND];
  __shared__ float r_s[KCAND];
  __shared__ int r_i[KCAND];
  int b = blockIdx.x;
  int tid = threadIdx.x;
  int gi = cidx[b * KCAND + tid];
  const float* bp = boxes + ((size_t)b * N + gi) * 4;
  float cx = bp[0], cy = bp[1], w = bp[2], h = bp[3];
  float sc = cscore[b * KCAND + tid];
  s_s[tid] = sc > CONF_THR ? sc : -1.f;
  s_cx[tid] = cx; s_cy[tid] = cy; s_w[tid] = w; s_h[tid] = h;
  float x1 = cx - w * 0.5f, y1 = cy - h * 0.5f;
  float x2 = cx + w * 0.5f, y2 = cy + h * 0.5f;
  s_x1[tid] = x1; s_y1[tid] = y1; s_x2[tid] = x2; s_y2[tid] = y2;
  s_ar[tid] = (x2 - x1) * (y2 - y1);
  __syncthreads();

  for (int it = 0; it < MAXDET; ++it) {
    r_s[tid] = s_s[tid];
    r_i[tid] = tid;
    __syncthreads();
    for (int off = 512; off > 0; off >>= 1) {
      if (tid < off) {
        float a = r_s[tid], c = r_s[tid + off];
        int ia = r_i[tid], ic = r_i[tid + off];
        if (c > a || (c == a && ic < ia)) { r_s[tid] = c; r_i[tid] = ic; }
      }
      __syncthreads();
    }
    int j = r_i[0];
    float smax = r_s[0];
    bool ok = smax > 0.f;
    if (ok) {
      float jx1 = s_x1[j], jy1 = s_y1[j], jx2 = s_x2[j], jy2 = s_y2[j], jar = s_ar[j];
      float iw = fmaxf(fminf(jx2, s_x2[tid]) - fmaxf(jx1, s_x1[tid]), 0.f);
      float ih = fmaxf(fminf(jy2, s_y2[tid]) - fmaxf(jy1, s_y1[tid]), 0.f);
      float inter = iw * ih;
      float uni = jar + s_ar[tid] - inter;
      float iou = inter / (uni + 1e-16f);
      if (iou > NMS_IOU || tid == j) s_s[tid] = -1.f;
    }
    if (tid < 4) {
      float v = 0.f;
      if (ok) v = (tid == 0) ? s_cx[j] : (tid == 1) ? s_cy[j] : (tid == 2) ? s_w[j] : s_h[j];
      out[((size_t)b * MAXDET + it) * 4 + tid] = v;
    }
    __syncthreads();
  }
}

// ---------------------------------------------------------------------------
extern "C" void kernel_launch(void* const* d_in, const int* in_sizes, int n_in,
                              void* d_out, int out_size, void* d_ws, size_t ws_size,
                              hipStream_t stream) {
  const float* features = (const float*)d_in[0];
  const float* w1 = (const float*)d_in[1];
  const float* b1 = (const float*)d_in[2];
  const float* g1 = (const float*)d_in[3];
  const float* be1 = (const float*)d_in[4];
  const float* m1 = (const float*)d_in[5];
  const float* v1 = (const float*)d_in[6];
  const float* w2 = (const float*)d_in[7];
  const float* b2 = (const float*)d_in[8];
  const float* g2 = (const float*)d_in[9];
  const float* be2 = (const float*)d_in[10];
  const float* m2 = (const float*)d_in[11];
  const float* v2 = (const float*)d_in[12];
  const float* wo = (const float*)d_in[13];
  const float* bo = (const float*)d_in[14];

  const int B = in_sizes[0] / (256 * HH * WW);
  char* ws = (char*)d_ws;
  size_t off = 0;
  auto take = [&](size_t bytes) {
    void* p = ws + off;
    off += (bytes + 255) & ~(size_t)255;
    return p;
  };
  const int S1 = 9 * 256 / 32, T1 = 128 / 16;   // 72, 8
  const int S2 = 9 * 128 / 32, T2 = 64 / 16;    // 36, 4
  unsigned int* wb1h = (unsigned int*)take((size_t)T1 * S1 * 256 * 4);
  unsigned int* wb1l = (unsigned int*)take((size_t)T1 * S1 * 256 * 4);
  float* scsh1 = (float*)take(2 * 128 * 4);
  unsigned int* wb2h = (unsigned int*)take((size_t)T2 * S2 * 256 * 4);
  unsigned int* wb2l = (unsigned int*)take((size_t)T2 * S2 * 256 * 4);
  float* scsh2 = (float*)take(2 * 64 * 4);
  float* x1 = (float*)take((size_t)B * 128 * HH * WW * 4);
  float* x2 = (float*)take((size_t)B * 64 * HH * WW * 4);
  float* boxes = (float*)take((size_t)B * HH * WW * NA * 4 * 4);
  float* scores = (float*)take((size_t)B * HH * WW * NA * 4);
  int* cidx = (int*)take((size_t)B * KCAND * 4);
  float* cscore = (float*)take((size_t)B * KCAND * 4);

  pack_weights<<<576, 256, 0, stream>>>(w1, b1, g1, be1, m1, v1, wb1h, wb1l, scsh1, 256, 128);
  pack_weights<<<144, 256, 0, stream>>>(w2, b2, g2, be2, m2, v2, wb2h, wb2l, scsh2, 128, 64);

  conv3x3_bn_relu<256, 128><<<B * HH * 4, 256, 0, stream>>>(features, wb1h, wb1l, scsh1, x1);
  conv3x3_bn_relu<128, 64><<<B * HH * 4, 128, 0, stream>>>(x1, wb2h, wb2l, scsh2, x2);

  int total = B * HH * WW * NA;
  head_decode<<<(total + 255) / 256, 256, 0, stream>>>(x2, wo, bo, boxes, scores, total);
  topk1024<<<B, 256, 0, stream>>>(scores, cidx, cscore);
  nms_kernel<<<B, 1024, 0, stream>>>(boxes, cidx, cscore, (float*)d_out);
}
